// FC_29970281791761
// MI455X (gfx1250) — matrix operand order repaired and checked by reading, not compiled, not run on hardware
//
#include <hip/hip_runtime.h>
#include <hip/hip_bf16.h>

typedef __bf16 bf16;
typedef __attribute__((ext_vector_type(16))) __bf16 v16bf;
typedef __attribute__((ext_vector_type(8)))  __bf16 v8bf;
typedef __attribute__((ext_vector_type(8)))  float   v8f;

#define BM 128
#define BN 128
#define BK 32
#define LDK 40   // padded LDS row stride (elements); multiples of 8 keep 16B alignment

// ---------------------------------------------------------------------------
// bf16 WMMA GEMM: C[M,N] = act(A[M,K] @ B[K,N] + bias), fp32 accumulate.
//   A  : [M,K]  row-major bf16, K % 32 == 0, rows 16B aligned
//   Bt : [N,K]  row-major bf16 (B pre-transposed), same K
// Block tile 128x128, K-step 32, 8 waves; wave tile 32x64 = 8 WMMA tiles.
// Double-buffered LDS stages; register prefetch of next K-tile.
// ---------------------------------------------------------------------------
__global__ __launch_bounds__(256)
void gemm_bf16_wmma(const bf16* __restrict__ A, const bf16* __restrict__ Bt,
                    const float* __restrict__ bias, float* __restrict__ C,
                    int M, int N, int K, int Nrows, int act)
{
    __shared__ __align__(16) bf16 As[2][BM * LDK];   // 2 x 10 KB
    __shared__ __align__(16) bf16 Bs[2][BN * LDK];   // 2 x 10 KB

    const int tid  = threadIdx.x;
    const int lane = tid & 31;
    const int wid  = tid >> 5;        // 0..7
    const int wm   = wid & 3;         // 32-row strip
    const int wn   = wid >> 2;        // 0..1 (64-col half)
    const int half = lane >> 4;       // 0/1
    const int l16  = lane & 15;
    const int m0   = blockIdx.y * BM;
    const int n0   = blockIdx.x * BN;

    v8f acc[2][4];
    #pragma unroll
    for (int a = 0; a < 2; ++a)
        #pragma unroll
        for (int b = 0; b < 4; ++b) acc[a][b] = v8f{};

    // prefetch registers: 2 chunks (8 bf16 each) per matrix per thread
    v8bf apre[2], bpre[2];

    auto fetch = [&](int kt) {
        #pragma unroll
        for (int i = 0; i < 2; ++i) {
            const int c  = tid + i * 256;     // 0..511
            const int r  = c >> 2;            // 0..127
            const int cc = (c & 3) << 3;      // 0,8,16,24
            v8bf av = {};
            const int gm = m0 + r;
            if (gm < M) av = *(const v8bf*)(A + (size_t)gm * K + kt + cc);
            apre[i] = av;
            v8bf bv = {};
            const int gn = n0 + r;
            if (gn < Nrows) bv = *(const v8bf*)(Bt + (size_t)gn * K + kt + cc);
            bpre[i] = bv;
        }
    };
    auto stage = [&](int s) {
        #pragma unroll
        for (int i = 0; i < 2; ++i) {
            const int c  = tid + i * 256;
            const int r  = c >> 2;
            const int cc = (c & 3) << 3;
            *(v8bf*)&As[s][r * LDK + cc] = apre[i];
            *(v8bf*)&Bs[s][r * LDK + cc] = bpre[i];
        }
    };

    fetch(0);
    stage(0);
    __syncthreads();

    int s = 0;
    for (int kt = 0; kt < K; kt += BK) {
        const bool more = (kt + BK) < K;
        if (more) fetch(kt + BK);             // overlaps with WMMA below

        // A fragments (2 sub-tiles of 16 rows), reused across 4 N-tiles
        v16bf af[2];
        #pragma unroll
        for (int sub = 0; sub < 2; ++sub) {
            const bf16* ap = &As[s][(wm * 32 + sub * 16 + l16) * LDK + half * 8];
            v8bf a0 = *(const v8bf*)ap;
            v8bf a1 = *(const v8bf*)(ap + 16);
            #pragma unroll
            for (int j = 0; j < 8; ++j) { af[sub][j] = a0[j]; af[sub][8 + j] = a1[j]; }
        }
        #pragma unroll
        for (int nt = 0; nt < 4; ++nt) {
            const bf16* bp = &Bs[s][(wn * 64 + nt * 16 + l16) * LDK + half * 8];
            v8bf b0 = *(const v8bf*)bp;
            v8bf b1 = *(const v8bf*)(bp + 16);
            v16bf bfrag;
            #pragma unroll
            for (int j = 0; j < 8; ++j) { bfrag[j] = b0[j]; bfrag[8 + j] = b1[j]; }
            acc[0][nt] = __builtin_amdgcn_wmma_f32_16x16x32_bf16(
                false, af[0], false, bfrag, (short)0, acc[0][nt], false, false);
            acc[1][nt] = __builtin_amdgcn_wmma_f32_16x16x32_bf16(
                false, af[1], false, bfrag, (short)0, acc[1][nt], false, false);
        }

        if (more) {
            stage(s ^ 1);
            __syncthreads();
            s ^= 1;
        }
    }

    // epilogue: bias + leaky, bounds-checked store
    #pragma unroll
    for (int nt = 0; nt < 4; ++nt) {
        const int n = n0 + wn * 64 + nt * 16 + l16;
        if (n < N) {
            const float bv = bias ? bias[n] : 0.0f;
            #pragma unroll
            for (int sub = 0; sub < 2; ++sub) {
                #pragma unroll
                for (int r = 0; r < 8; ++r) {
                    const int m = m0 + wm * 32 + sub * 16 + half * 8 + r;
                    if (m < M) {
                        float v = acc[sub][nt][r] + bv;
                        if (act) v = v > 0.0f ? v : 0.01f * v;
                        C[(size_t)m * N + n] = v;
                    }
                }
            }
        }
    }
}

// ---------------------------------------------------------------------------
// f32 -> bf16, zero-padded columns to Cp (A-side operands)
// ---------------------------------------------------------------------------
__global__ void cvt_pad_bf16(const float* __restrict__ src, bf16* __restrict__ dst,
                             int R, int C, int Cp, int total)
{
    int i = blockIdx.x * blockDim.x + threadIdx.x;
    if (i >= total) return;
    int r = i / Cp, c = i % Cp;
    float v = (r < R && c < C) ? src[(size_t)r * C + c] : 0.0f;
    dst[i] = (bf16)v;
}

// f32 [K,N] -> bf16 transposed [N,Kp], zero-padded K (B-side operands)
__global__ void cvt_pad_bf16_t(const float* __restrict__ src, bf16* __restrict__ dst,
                               int K, int N, int Kp, int total)
{
    int i = blockIdx.x * blockDim.x + threadIdx.x;
    if (i >= total) return;
    int n = i / Kp, k = i % Kp;
    float v = (k < K) ? src[(size_t)k * N + n] : 0.0f;
    dst[i] = (bf16)v;
}

__global__ void zero_f32(float* __restrict__ p, int n)
{
    int i = blockIdx.x * blockDim.x + threadIdx.x;
    if (i < n) p[i] = 0.0f;
}

__global__ void deg_accum(const int* __restrict__ col, const float* __restrict__ ew,
                          float* __restrict__ deg, int E)
{
    int e = blockIdx.x * blockDim.x + threadIdx.x;
    if (e < E) atomicAdd(&deg[col[e]], ew[e]);
}

__global__ void deg_finalize(float* __restrict__ deg, int n)
{
    int i = blockIdx.x * blockDim.x + threadIdx.x;
    if (i < n) {
        float d = deg[i] + 1.0f;            // + self-loop weight 1
        deg[i] = d > 0.0f ? rsqrtf(d) : 0.0f;
    }
}

__global__ void self_loop_init(const float* __restrict__ h, const float* __restrict__ dinv,
                               float* __restrict__ agg, int total, int F)
{
    int i = blockIdx.x * blockDim.x + threadIdx.x;
    if (i < total) {
        float d = dinv[i / F];
        agg[i] = h[i] * d * d;
    }
}

__global__ __launch_bounds__(256)
void edge_scatter(const int* __restrict__ row, const int* __restrict__ col,
                  const float* __restrict__ ew, const float* __restrict__ dinv,
                  const float* __restrict__ h, float* __restrict__ agg, int E, int F)
{
    int e = blockIdx.x;
    if (e >= E) return;
    int r = row[e], c = col[e];
    float norm = dinv[r] * ew[e] * dinv[c];
    const float* hr = h + (size_t)r * F;
    float* ac = agg + (size_t)c * F;
    for (int f = threadIdx.x; f < F; f += blockDim.x)
        atomicAdd(&ac[f], hr[f] * norm);
}

__global__ void bias_leaky_gather(const float* __restrict__ agg, const float* __restrict__ bias,
                                  const int* __restrict__ idx, float* __restrict__ feat,
                                  int Bsz, int F, int FEAT, int off)
{
    int i = blockIdx.x * blockDim.x + threadIdx.x;
    if (i >= Bsz * F) return;
    int b = i / F, f = i % F;
    float v = agg[(size_t)idx[b] * F + f] + bias[f];
    v = v > 0.0f ? v : 0.01f * v;
    feat[(size_t)b * FEAT + off + f] = v;
}

__global__ void copy_cols(const float* __restrict__ src, float* __restrict__ feat,
                          int Bsz, int F, int FEAT, int off)
{
    int i = blockIdx.x * blockDim.x + threadIdx.x;
    if (i >= Bsz * F) return;
    int b = i / F, f = i % F;
    feat[(size_t)b * FEAT + off + f] = src[(size_t)b * F + f];
}

// BN(eval) + leaky + dot with out_W2 + out_b2; one wave32 per row.
__global__ __launch_bounds__(256)
void bn_head(const float* __restrict__ h, const float* __restrict__ g,
             const float* __restrict__ be, const float* __restrict__ mu,
             const float* __restrict__ var, const float* __restrict__ W2,
             const float* __restrict__ b2, float* __restrict__ y, int Bsz)
{
    int wid = threadIdx.x >> 5, lane = threadIdx.x & 31;
    int m = blockIdx.x * 8 + wid;
    if (m >= Bsz) return;
    float s = 0.0f;
    for (int c = lane; c < 128; c += 32) {
        float v = (h[(size_t)m * 128 + c] - mu[c]) * rsqrtf(var[c] + 1e-5f) * g[c] + be[c];
        v = v > 0.0f ? v : 0.01f * v;
        s += v * W2[c];
    }
    #pragma unroll
    for (int off = 16; off > 0; off >>= 1) s += __shfl_xor(s, off, 32);
    if (lane == 0) y[m] = s + b2[0];
}

// ---------------------------------------------------------------------------
extern "C" void kernel_launch(void* const* d_in, const int* in_sizes, int n_in,
                              void* d_out, int out_size, void* d_ws, size_t ws_size,
                              hipStream_t stream)
{
    (void)in_sizes; (void)n_in; (void)out_size; (void)ws_size;

    const int Bsz = 4096, ND = 20000, NP = 10000, DE = 320000, PE = 160000;
    const int F = 1024, FEAT = 3372;
    const int KP_P = 2816;   // pad(2812, 32)
    const int KP_F = 3392;   // pad(3372, 32)

    const int*   d_index  = (const int*)  d_in[0];
    const int*   p_index  = (const int*)  d_in[1];
    const float* d_vecs   = (const float*)d_in[2];
    const float* p_emb    = (const float*)d_in[3];
    const float* d_ecfps  = (const float*)d_in[4];
    const int*   d_eidx   = (const int*)  d_in[5];
    const float* d_ew     = (const float*)d_in[6];
    const float* p_gos    = (const float*)d_in[7];
    const int*   p_eidx   = (const int*)  d_in[8];
    const float* p_ew     = (const float*)d_in[9];
    const float* d_gcn_W  = (const float*)d_in[10];
    const float* d_gcn_b  = (const float*)d_in[11];
    const float* p_gcn_W  = (const float*)d_in[12];
    const float* p_gcn_b  = (const float*)d_in[13];
    const float* enc_W1   = (const float*)d_in[14];
    const float* enc_b1   = (const float*)d_in[15];
    const float* enc_W2   = (const float*)d_in[16];
    const float* enc_b2   = (const float*)d_in[17];
    const float* dec_W1   = (const float*)d_in[18];
    const float* dec_b1   = (const float*)d_in[19];
    const float* dec_W2   = (const float*)d_in[20];
    const float* dec_b2   = (const float*)d_in[21];
    const float* out_W1   = (const float*)d_in[22];
    const float* out_b1   = (const float*)d_in[23];
    const float* bn_g     = (const float*)d_in[24];
    const float* bn_be    = (const float*)d_in[25];
    const float* bn_mu    = (const float*)d_in[26];
    const float* bn_var   = (const float*)d_in[27];
    const float* out_W2   = (const float*)d_in[28];
    const float* out_b2   = (const float*)d_in[29];

    // outputs: y | encoded | decoded | feature
    float* y       = (float*)d_out;
    float* encoded = y + 4096;
    float* decoded = encoded + (size_t)Bsz * 256;
    float* feature = decoded + (size_t)Bsz * FEAT;

    // workspace layout (256 B aligned regions)
    char* ws = (char*)d_ws;
    size_t off = 0;
    auto take = [&](size_t bytes) { char* p = ws + off; off = (off + bytes + 255) & ~(size_t)255; return p; };
    bf16*  XBF   = (bf16*) take((size_t)28160000 * 2);            // max A matrix
    bf16*  WBF   = (bf16*) take((size_t)KP_F * 1024 * 2);         // max B^T matrix
    float* H     = (float*)take((size_t)ND * F * 4);              // GEMM outputs
    float* AGG   = (float*)take((size_t)ND * F * 4);              // GCN aggregate
    float* DEG   = (float*)take((size_t)ND * 4);
    bf16*  HBF   = (bf16*) take((size_t)Bsz * 1024 * 2);          // hidden bf16
    bf16*  ENCBF = (bf16*) take((size_t)Bsz * 256 * 2);           // encoded bf16

    auto cvt = [&](const float* src, bf16* dst, int R, int C, int Cp) {
        int total = R * Cp;
        cvt_pad_bf16<<<(total + 255) / 256, 256, 0, stream>>>(src, dst, R, C, Cp, total);
    };
    // B [K,N] -> B^T [N,Kp]
    auto cvtT = [&](const float* src, bf16* dst, int K, int N, int Kp) {
        int total = N * Kp;
        cvt_pad_bf16_t<<<(total + 255) / 256, 256, 0, stream>>>(src, dst, K, N, Kp, total);
    };
    auto gemm = [&](const bf16* A, const bf16* BT, const float* bias, float* Cm,
                    int M, int N, int K, int act) {
        dim3 g((N + BN - 1) / BN, (M + BM - 1) / BM);
        gemm_bf16_wmma<<<g, 256, 0, stream>>>(A, BT, bias, Cm, M, N, K, N, act);
    };

    // ---------------- GCN branch helper ----------------
    auto gcn = [&](const float* X, const float* W, const float* bias, const int* eidx,
                   const float* ew, const int* gather_idx, int n, int E, int Kin, int Kp,
                   int feat_off) {
        // h = X @ W
        cvt(X, XBF, n, Kin, Kp);
        cvtT(W, WBF, Kin, F, Kp);
        gemm(XBF, WBF, nullptr, H, n, F, Kp, 0);
        // degree / dinv
        zero_f32<<<(n + 255) / 256, 256, 0, stream>>>(DEG, n);
        deg_accum<<<(E + 255) / 256, 256, 0, stream>>>(eidx + E, ew, DEG, E);
        deg_finalize<<<(n + 255) / 256, 256, 0, stream>>>(DEG, n);
        // aggregate: self loop + edges
        int total = n * F;
        self_loop_init<<<(total + 255) / 256, 256, 0, stream>>>(H, DEG, AGG, total, F);
        edge_scatter<<<E, 256, 0, stream>>>(eidx, eidx + E, ew, DEG, H, AGG, E, F);
        // bias + leaky + gather into feature block
        bias_leaky_gather<<<(Bsz * F + 255) / 256, 256, 0, stream>>>(
            AGG, bias, gather_idx, feature, Bsz, F, FEAT, feat_off);
    };

    // drug branch -> feature[:, 1324:2348]
    gcn(d_ecfps, d_gcn_W, d_gcn_b, d_eidx, d_ew, d_index, ND, DE, 1024, 1024, 1324);
    // protein branch -> feature[:, 2348:3372]
    gcn(p_gos, p_gcn_W, p_gcn_b, p_eidx, p_ew, p_index, NP, PE, 2812, KP_P, 2348);

    // feature[:, 0:300] = d_vecs ; feature[:, 300:1324] = p_embeddings
    copy_cols<<<(Bsz * 300 + 255) / 256, 256, 0, stream>>>(d_vecs, feature, Bsz, 300, FEAT, 0);
    copy_cols<<<(Bsz * 1024 + 255) / 256, 256, 0, stream>>>(p_emb, feature, Bsz, 1024, FEAT, 300);

    // ---------------- encoder ----------------
    cvt(feature, XBF, Bsz, FEAT, KP_F);                    // [4096, 3392] bf16
    cvtT(enc_W1, WBF, FEAT, 1024, KP_F);                   // [1024, 3392]
    gemm(XBF, WBF, enc_b1, H, Bsz, 1024, KP_F, 1);         // leaky
    cvt(H, HBF, Bsz, 1024, 1024);
    cvtT(enc_W2, WBF, 1024, 256, 1024);                    // [256, 1024]
    gemm(HBF, WBF, enc_b2, encoded, Bsz, 256, 1024, 1);    // leaky -> d_out
    cvt(encoded, ENCBF, Bsz, 256, 256);

    // ---------------- decoder ----------------
    cvtT(dec_W1, WBF, 256, 1024, 256);                     // [1024, 256]
    gemm(ENCBF, WBF, dec_b1, H, Bsz, 1024, 256, 1);        // leaky
    cvt(H, HBF, Bsz, 1024, 1024);
    cvtT(dec_W2, WBF, 1024, FEAT, 1024);                   // [3372, 1024]
    gemm(HBF, WBF, dec_b2, decoded, Bsz, FEAT, 1024, 1);   // leaky -> d_out

    // ---------------- head ----------------
    cvtT(out_W1, WBF, 256, 128, 256);                      // [128, 256]
    gemm(ENCBF, WBF, out_b1, H, Bsz, 128, 256, 0);         // no act (BN follows)
    bn_head<<<(Bsz + 7) / 8, 256, 0, stream>>>(H, bn_g, bn_be, bn_mu, bn_var,
                                               out_W2, out_b2, y, Bsz);
}
